// CrossViewPhotoDepthSinglePose_37855841747695
// MI455X (gfx1250) — matrix operand order repaired and checked by reading, not compiled, not run on hardware
//
#include <hip/hip_runtime.h>
#include <hip/hip_bf16.h>
#include <math.h>

// ---------------- problem constants ----------------
#define B_   2
#define V_   6
#define H_   256
#define W_   384
#define C_   3
#define P_   30            // V*(V-1) ordered pairs
#define BP_  (P_*B_)       // 60 pair-batches
#define HW_  (H_*W_)
#define HC_  246           // H - 10 (valid conv)
#define WC_  374           // W - 10
#define MIN_D 0.001f
#define MAX_D 80.0f
#define SSIM_W_ 0.85f

// workspace layout (float offsets)
#define OFF_CAM   0                         // B*V*16 = 192
#define OFF_NACC  192                       // 30
#define OFF_SQACC 222                       // 30
#define OFF_ADACC 252                       // 30
#define OFF_SSACC 282                       // 30
#define OFF_WDEP  320                       // BP*H*W
#define OFF_WIMG  (320 + BP_*HW_)           // BP*3*H*W

#if __has_builtin(__builtin_amdgcn_global_load_async_to_lds_b32) && \
    __has_builtin(__builtin_amdgcn_s_wait_asynccnt)
#define USE_ASYNC_LDS 1
#else
#define USE_ASYNC_LDS 0
#endif

typedef __attribute__((address_space(1))) int g_i32;   // global addrspace
typedef __attribute__((address_space(3))) int l_i32;   // LDS addrspace

typedef __attribute__((ext_vector_type(16))) _Float16 v16h;
typedef __attribute__((ext_vector_type(8)))  float    v8f;

// 11-tap Gaussian, sigma=1.5, normalized (matches reference _gauss1d)
__device__ __constant__ float c_g[11] = {
    0.00102838f, 0.00759878f, 0.03600075f, 0.10936069f, 0.21300554f,
    0.26601173f, 0.21300554f, 0.10936069f, 0.03600075f, 0.00759878f,
    0.00102838f };

struct Cam { float R[9]; float T[3]; float fx, fy, cx, cy; };

__device__ inline Cam loadCam(const float* ws, int b, int v) {
    Cam c;
    const float* q = ws + OFF_CAM + (b * V_ + v) * 16;
#pragma unroll
    for (int i = 0; i < 9; ++i) c.R[i] = q[i];
    c.T[0] = q[9]; c.T[1] = q[10]; c.T[2] = q[11];
    c.fx = q[12]; c.fy = q[13]; c.cx = q[14]; c.cy = q[15];
    return c;
}

__device__ inline void pairViews(int p, int& tv, int& sv) {
    tv = p / (V_ - 1);
    int r = p % (V_ - 1);
    sv = r + (r >= tv ? 1 : 0);
}

__device__ inline float clip01(float x) { return fminf(fmaxf(x, 0.f), 1.f); }

// ---------------- kernel 1: poses + zero accumulators ----------------
__global__ void k_pose(const float* __restrict__ pose, float* __restrict__ ws) {
    int t = threadIdx.x;
    if (t < P_) {
        ws[OFF_NACC + t]  = 0.f;
        ws[OFF_SQACC + t] = 0.f;
        ws[OFF_ADACC + t] = 0.f;
        ws[OFF_SSACC + t] = 0.f;
    }
    if (t < B_ * V_) {
        const float* pe = pose + t * 9;
        float qr = pe[3], qi = pe[4], qj = pe[5], qk = pe[6];
        float s = 2.f / (qr*qr + qi*qi + qj*qj + qk*qk);
        float* q = ws + OFF_CAM + t * 16;
        q[0] = 1.f - s*(qj*qj + qk*qk);
        q[1] = s*(qi*qj - qk*qr);
        q[2] = s*(qi*qk + qj*qr);
        q[3] = s*(qi*qj + qk*qr);
        q[4] = 1.f - s*(qi*qi + qk*qk);
        q[5] = s*(qj*qk - qi*qr);
        q[6] = s*(qi*qk - qj*qr);
        q[7] = s*(qj*qk + qi*qr);
        q[8] = 1.f - s*(qi*qi + qj*qj);
        q[9]  = pe[0]; q[10] = pe[1]; q[11] = pe[2];
        float fovh = pe[7], fovw = pe[8];
        q[12] = ((float)W_ * 0.5f) / tanf(fovw * 0.5f);   // fx
        q[13] = ((float)H_ * 0.5f) / tanf(fovh * 0.5f);   // fy
        q[14] = (float)W_ * 0.5f;                         // cx
        q[15] = (float)H_ * 0.5f;                         // cy
    }
}

// ---------------- kernel 2: init scattered-depth buffer to +inf ----------------
__global__ void k_initdep(float* __restrict__ ws) {
    int i = blockIdx.x * 256 + threadIdx.x;
    ws[OFF_WDEP + i] = __int_as_float(0x7f800000);  // +inf
}

// ---------------- kernel 3: scatter-min depth reprojection ----------------
__global__ __launch_bounds__(256)
void k_scatter(const float* __restrict__ depth, float* __restrict__ ws) {
    int l  = blockIdx.x * 256 + threadIdx.x;   // pixel
    int pb = blockIdx.y;
    int p = pb / B_, b = pb % B_;
    int tv, sv; pairViews(p, tv, sv);
    Cam ct = loadCam(ws, b, tv);
    Cam cs = loadCam(ws, b, sv);
    int h = l / W_, w = l % W_;

    float ds = depth[((b * V_ + sv) * H_ + h) * W_ + w];
    float qz = fminf(fmaxf(ds, MIN_D), MAX_D);
    float qx = ((float)w - cs.cx) * qz / (cs.fx + 1e-8f);
    float qy = ((float)h - cs.cy) * qz / (cs.fy + 1e-8f);
    float ax = qx - cs.T[0], ay = qy - cs.T[1], az = qz - cs.T[2];
    // world2 = Rs^T * a
    float wx = cs.R[0]*ax + cs.R[3]*ay + cs.R[6]*az;
    float wy = cs.R[1]*ax + cs.R[4]*ay + cs.R[7]*az;
    float wz = cs.R[2]*ax + cs.R[5]*ay + cs.R[8]*az;
    // cam_t = Rt * world2 + tt
    float cx0 = ct.R[0]*wx + ct.R[1]*wy + ct.R[2]*wz + ct.T[0];
    float cy0 = ct.R[3]*wx + ct.R[4]*wy + ct.R[5]*wz + ct.T[1];
    float cz0 = ct.R[6]*wx + ct.R[7]*wy + ct.R[8]*wz + ct.T[2];
    float zt  = fmaxf(cz0, 1e-4f);
    float ut  = ct.fx * cx0 / zt + ct.cx;
    float vt  = ct.fy * cy0 / zt + ct.cy;
    if (zt > 1e-4f) {
        float uf = rintf(ut), vf = rintf(vt);
        if (uf >= 0.f && uf <= (float)(W_ - 1) && vf >= 0.f && vf <= (float)(H_ - 1)) {
            int ui = (int)uf, vi = (int)vf;
            atomicMin((unsigned int*)&ws[OFF_WDEP + pb * HW_ + vi * W_ + ui],
                      __float_as_uint(zt));
        }
    }
}

// ---------------- kernel 4: warp + masked L2 / L1 accumulation ----------------
__global__ __launch_bounds__(256)
void k_warp(const float* __restrict__ depth, const float* __restrict__ cpred,
            const float* __restrict__ cgt, const unsigned char* __restrict__ vmask,
            float* __restrict__ ws) {
    __shared__ float sN, sSq, sAd;
    if (threadIdx.x == 0) { sN = 0.f; sSq = 0.f; sAd = 0.f; }
    __syncthreads();

    int l  = blockIdx.x * 256 + threadIdx.x;
    int pb = blockIdx.y;
    int p = pb / B_, b = pb % B_;
    int tv, sv; pairViews(p, tv, sv);
    Cam ct = loadCam(ws, b, tv);
    Cam cs = loadCam(ws, b, sv);
    int h = l / W_, w = l % W_;

    __builtin_prefetch(&ws[OFF_WDEP + pb * HW_ + l], 0, 1);  // global_prefetch_b8

    float dt = depth[((b * V_ + tv) * H_ + h) * W_ + w];
    float px = ((float)w - ct.cx) * dt / ct.fx;
    float py = ((float)h - ct.cy) * dt / ct.fy;
    float pz = dt;
    float ax = px - ct.T[0], ay = py - ct.T[1], az = pz - ct.T[2];
    // world = Rt^T * a
    float wx = ct.R[0]*ax + ct.R[3]*ay + ct.R[6]*az;
    float wy = ct.R[1]*ax + ct.R[4]*ay + ct.R[7]*az;
    float wz = ct.R[2]*ax + ct.R[5]*ay + ct.R[8]*az;
    // cam_s = Rs * world + ts
    float sx = cs.R[0]*wx + cs.R[1]*wy + cs.R[2]*wz + cs.T[0];
    float sy = cs.R[3]*wx + cs.R[4]*wy + cs.R[5]*wz + cs.T[1];
    float szr = cs.R[6]*wx + cs.R[7]*wy + cs.R[8]*wz + cs.T[2];
    float sz = fmaxf(szr, 1e-4f);
    float us = cs.fx * sx / sz + cs.cx;
    float vs = cs.fy * sy / sz + cs.cy;

    float x0f = floorf(us), y0f = floorf(vs);
    float fxw = us - x0f, fyw = vs - y0f;
    bool inb = (us >= 0.f) && (us <= (float)(W_ - 1)) &&
               (vs >= 0.f) && (vs <= (float)(H_ - 1));
    bool mimg = inb && (szr > 1e-4f);
    float mf = mimg ? 1.f : 0.f;

    int xi0 = (int)fminf(fmaxf(x0f,       0.f), (float)(W_ - 1));
    int yi0 = (int)fminf(fmaxf(y0f,       0.f), (float)(H_ - 1));
    int xi1 = (int)fminf(fmaxf(x0f + 1.f, 0.f), (float)(W_ - 1));
    int yi1 = (int)fminf(fmaxf(y0f + 1.f, 0.f), (float)(H_ - 1));
    float w00 = (1.f - fxw) * (1.f - fyw);
    float w10 = fxw * (1.f - fyw);
    float w01 = (1.f - fxw) * fyw;
    float w11 = fxw * fyw;

    int srcb = ((b * V_ + sv) * 3) * HW_;
    int tgtb = ((b * V_ + tv) * 3) * HW_;
    int wib  = (pb * 3) * HW_;
    float dsq = 0.f;
#pragma unroll
    for (int c = 0; c < 3; ++c) {
        const float* sp = cpred + srcb + c * HW_;
        float g00 = clip01(sp[yi0 * W_ + xi0]);
        float g10 = clip01(sp[yi0 * W_ + xi1]);
        float g01 = clip01(sp[yi1 * W_ + xi0]);
        float g11 = clip01(sp[yi1 * W_ + xi1]);
        float wv = (g00 * w00 + g10 * w10 + g01 * w01 + g11 * w11) * mf;
        ws[OFF_WIMG + wib + c * HW_ + l] = wv;
        float itc = clip01((cgt[tgtb + c * HW_ + l] + 1.f) * 0.5f);
        float d = wv - itc;
        dsq += d * d;
    }

    float wdv = ws[OFF_WDEP + pb * HW_ + l];
    bool mdep = (wdv < __int_as_float(0x7f800000));
    float wdf = mdep ? wdv : 0.f;
    bool vm = vmask[((b * V_ + tv) * H_ + h) * W_ + w] != 0;
    bool va = vm && mimg && mdep &&
              (dt > MIN_D) && (dt < MAX_D) && (wdf > MIN_D) && (wdf < MAX_D);

    // wave32 shuffle reduction, then one LDS atomic per wave per quantity
    float vN  = va ? 1.f : 0.f;
    float vSq = va ? dsq : 0.f;
    float vAd = va ? fabsf(dt - wdf) : 0.f;
#pragma unroll
    for (int off = 16; off > 0; off >>= 1) {
        vN  += __shfl_down(vN,  off, 32);
        vSq += __shfl_down(vSq, off, 32);
        vAd += __shfl_down(vAd, off, 32);
    }
    if ((threadIdx.x & 31) == 0 && vN > 0.f) {
        atomicAdd(&sN, vN);
        atomicAdd(&sSq, vSq);
        atomicAdd(&sAd, vAd);
    }
    __syncthreads();
    if (threadIdx.x == 0) {
        atomicAdd(&ws[OFF_NACC + p],  sN);
        atomicAdd(&ws[OFF_SQACC + p], sSq);
        atomicAdd(&ws[OFF_ADACC + p], sAd);
    }
}

// ---------------- kernel 5: fused SSIM via WMMA ----------------
// one wave per 16x16 output tile; separable 11-tap blur as two 16x16x32 f16 WMMAs
// (vertical Gv*X, horizontal T*Gh) for each of 5 maps {x, y, xx, yy, xy}.
__device__ inline v8f wmma16(v16h a, v16h b, v8f c) {
    return __builtin_amdgcn_wmma_f32_16x16x32_f16(false, a, false, b,
                                                  (short)0, c, false, false);
}

__device__ inline float mapval(int m, float x, float y) {
    switch (m) {
        case 0: return x;
        case 1: return y;
        case 2: return x * x;
        case 3: return y * y;
        default: return x * y;
    }
}

#define TILES_X 24
#define TILES_Y 16
#define TILES_PER 384   // TILES_X*TILES_Y

__global__ __launch_bounds__(128)
void k_ssim(const float* __restrict__ cgt, float* __restrict__ ws) {
    __shared__ float Xs[4][32][33];
    __shared__ float Ys[4][32][33];
    __shared__ float Ts[4][16][33];

    const int wid  = threadIdx.x >> 5;
    const int lane = threadIdx.x & 31;
    int g = blockIdx.x * 4 + wid;            // grid sized so every wave has work

    int img  = g / TILES_PER;                // (pb,c) index, 0..179
    int tile = g % TILES_PER;
    int ty = tile / TILES_X, tx = tile % TILES_X;
    int c  = img % 3;
    int pb = img / 3;
    int p = pb / B_, b = pb % B_;
    int tv = p / (V_ - 1);
    int r0 = ty * 16, c0 = tx * 16;

    const float* wimg = ws + OFF_WIMG + (pb * 3 + c) * HW_;
    const float* gtp  = cgt + ((b * V_ + tv) * 3 + c) * HW_;

    // stage the 32x32 (clamped) input region into LDS.
    // X plane holds RAW gt (transformed in-place below); Y plane holds wimg.
#if USE_ASYNC_LDS
#pragma unroll 4
    for (int i = 0; i < 32; ++i) {
        int gr = min(r0 + i,    H_ - 1);
        int gc = min(c0 + lane, W_ - 1);
        __builtin_amdgcn_global_load_async_to_lds_b32(
            (g_i32*)(gtp + gr * W_ + gc),
            (l_i32*)&Xs[wid][i][lane], 0, 0);
        __builtin_amdgcn_global_load_async_to_lds_b32(
            (g_i32*)(wimg + gr * W_ + gc),
            (l_i32*)&Ys[wid][i][lane], 0, 0);
    }
    __builtin_amdgcn_s_wait_asynccnt(0);
#else
#pragma unroll 4
    for (int i = 0; i < 32; ++i) {
        int gr = min(r0 + i,    H_ - 1);
        int gc = min(c0 + lane, W_ - 1);
        Xs[wid][i][lane] = gtp[gr * W_ + gc];
        Ys[wid][i][lane] = wimg[gr * W_ + gc];
    }
#endif
    // in-place transform of gt: it = clip((gt+1)/2, 0, 1)  (per-wave slice)
#pragma unroll 4
    for (int i = 0; i < 32; ++i) {
        Xs[wid][i][lane] = clip01((Xs[wid][i][lane] + 1.f) * 0.5f);
    }
    __syncthreads();

    const int mrow = lane & 15;
    const int lo8  = (lane < 16) ? 0 : 8;
    const int kb   = (lane < 16) ? 0 : 16;
    const int nn   = lane & 15;

    // A operand: vertical Gaussian band matrix Gv[16x32], A[M][K]=g[K-M]
    v16h av;
#pragma unroll
    for (int hh = 0; hh < 16; ++hh) {
        int K = (hh < 8) ? (lo8 + hh) : (16 + lo8 + hh - 8);
        int d = K - mrow;
        av[hh] = (d >= 0 && d <= 10) ? (_Float16)c_g[d] : (_Float16)0.0f;
    }
    // B operand: horizontal Gaussian band matrix Gh[32x16], B[K][N]=g[K-N]
    v16h bh;
#pragma unroll
    for (int hh = 0; hh < 16; ++hh) {
        int K = (hh < 8) ? (lo8 + hh) : (16 + lo8 + hh - 8);
        int d = K - nn;
        bh[hh] = (d >= 0 && d <= 10) ? (_Float16)c_g[d] : (_Float16)0.0f;
    }

    v8f res[5];
#pragma unroll
    for (int m = 0; m < 5; ++m) {
        // vertical pass: two WMMAs cover columns 0..15 and 16..31
        v16h bv0, bv1;
#pragma unroll
        for (int hh = 0; hh < 16; ++hh) {
            int K = (hh < 8) ? (lo8 + hh) : (16 + lo8 + hh - 8);  // region row
            float x0 = Xs[wid][K][nn],      y0 = Ys[wid][K][nn];
            float x1 = Xs[wid][K][16 + nn], y1 = Ys[wid][K][16 + nn];
            bv0[hh] = (_Float16)mapval(m, x0, y0);
            bv1[hh] = (_Float16)mapval(m, x1, y1);
        }
        v8f cz = {};
        v8f d0 = wmma16(av, bv0, cz);
        v8f d1 = wmma16(av, bv1, cz);
#pragma unroll
        for (int j = 0; j < 8; ++j) {
            int M = (lane < 16) ? j : (j + 8);
            Ts[wid][M][nn]      = d0[j];
            Ts[wid][M][16 + nn] = d1[j];
        }
        __syncthreads();
        // horizontal pass: A = T (f32 -> f16), B = Gh
        v16h ah;
#pragma unroll
        for (int hh = 0; hh < 16; ++hh) {
            int K = (hh < 8) ? (lo8 + hh) : (16 + lo8 + hh - 8);
            ah[hh] = (_Float16)Ts[wid][mrow][K];
        }
        res[m] = wmma16(ah, bh, cz);
        __syncthreads();
    }

    // SSIM map + masked reduction
    const float C1v = 1e-4f, C2v = 9e-4f;
    float part = 0.f;
#pragma unroll
    for (int j = 0; j < 8; ++j) {
        int row = r0 + ((lane < 16) ? j : (j + 8));
        int col = c0 + (lane & 15);
        if (row < HC_ && col < WC_) {
            float mu1 = res[0][j], mu2 = res[1][j];
            float bxx = res[2][j], byy = res[3][j], bxy = res[4][j];
            float mu11 = mu1 * mu1, mu22 = mu2 * mu2, mu12 = mu1 * mu2;
            float s1 = bxx - mu11, s2 = byy - mu22, s12 = bxy - mu12;
            float sm = ((2.f * mu12 + C1v) * (2.f * s12 + C2v)) /
                       ((mu11 + mu22 + C1v) * (s1 + s2 + C2v));
            part += sm;
        }
    }
#pragma unroll
    for (int off = 16; off > 0; off >>= 1)
        part += __shfl_down(part, off, 32);
    if (lane == 0) atomicAdd(&ws[OFF_SSACC + p], part);
}

// ---------------- kernel 6: finalize ----------------
__global__ void k_final(const float* __restrict__ ws, float* __restrict__ out) {
    if (threadIdx.x != 0 || blockIdx.x != 0) return;
    const float ssim_cnt = (float)(B_ * 3 * HC_ * WC_);
    float tps = 0.f, tds = 0.f, np = 0.f;
    for (int p = 0; p < P_; ++p) {
        float n  = ws[OFF_NACC + p];
        float sm = ws[OFF_SSACC + p] / ssim_cnt;
        float l2 = ws[OFF_SQACC + p] / fmaxf(3.f * n, 1.f);
        float photo = SSIM_W_ * (1.f - sm) + (1.f - SSIM_W_) * l2;
        float dl = ws[OFF_ADACC + p] / fmaxf(n, 1.f);
        float has = (n > 0.f) ? 1.f : 0.f;
        tps += has * photo;
        tds += has * dl;
        np  += has;
    }
    float inv = (np > 0.f) ? (1.f / fmaxf(np, 1.f)) : 0.f;
    float lp = tps * inv;   // W_PHOTO = 1
    float ld = tds * inv;   // W_DEPTH = 1
    out[0] = lp;
    out[1] = ld;
    float s = lp + ld;
    out[2] = isfinite(s) ? s : 0.f;
}

// ---------------- host launcher ----------------
extern "C" void kernel_launch(void* const* d_in, const int* in_sizes, int n_in,
                              void* d_out, int out_size, void* d_ws, size_t ws_size,
                              hipStream_t stream) {
    const float* pose  = (const float*)d_in[0];
    const float* depth = (const float*)d_in[1];
    const float* cpred = (const float*)d_in[2];
    const float* cgt   = (const float*)d_in[3];
    const unsigned char* vmask = (const unsigned char*)d_in[4];
    float* out = (float*)d_out;
    float* ws  = (float*)d_ws;

    (void)in_sizes; (void)n_in; (void)out_size; (void)ws_size;

    k_pose<<<1, 64, 0, stream>>>(pose, ws);
    k_initdep<<<(BP_ * HW_) / 256, 256, 0, stream>>>(ws);
    k_scatter<<<dim3(HW_ / 256, BP_), 256, 0, stream>>>(depth, ws);
    k_warp<<<dim3(HW_ / 256, BP_), 256, 0, stream>>>(depth, cpred, cgt, vmask, ws);
    k_ssim<<<(BP_ * 3 * TILES_PER) / 4, 128, 0, stream>>>(cgt, ws);
    k_final<<<1, 32, 0, stream>>>(ws, out);
}
